// Transformer_33449205301304
// MI455X (gfx1250) — matrix operand order repaired and checked by reading, not compiled, not run on hardware
//
#include <hip/hip_runtime.h>
#include <hip/hip_bf16.h>

#define NUM_LAYERS 4
#define D_MODEL 1024
#define D_FF 4096
#define SEQ 2048
#define N_HEADS 16
#define HD 64
#define KCONV 7

typedef unsigned short u16;
typedef __attribute__((ext_vector_type(16))) __bf16 v16bf;
typedef __attribute__((ext_vector_type(8)))  float  v8f;
typedef __attribute__((ext_vector_type(4)))  unsigned int v4u;
typedef __attribute__((ext_vector_type(4)))  unsigned int u32x4;
typedef __attribute__((ext_vector_type(8)))  int i32x8;
typedef __attribute__((ext_vector_type(4)))  int i32x4;

union Frag { v4u u4[2]; v16bf v; };

#if defined(__has_builtin)
#if __has_builtin(__builtin_amdgcn_tensor_load_to_lds)
#define HAVE_TDM 1
#endif
#endif
#ifndef HAVE_TDM
#define HAVE_TDM 0
#endif
#if HAVE_TDM && (__clang_major__ >= 23)
#define TDM_6ARG 1
#else
#define TDM_6ARG 0
#endif

static __device__ __forceinline__ u16 f2bf(float f) {
    union { float f; unsigned int u; } v; v.f = f;
    unsigned int r = v.u + 0x7FFFu + ((v.u >> 16) & 1u);   // round-to-nearest-even
    return (u16)(r >> 16);
}

#if HAVE_TDM
// ---------------------------------------------------------------------------
// TDM 2-D tile load: global (row-major, stride_elems) -> LDS, bf16 elements.
// Descriptor packing per CDNA5 ISA sec 8.3/8.4 (D# group0/group1).
// ---------------------------------------------------------------------------
static __device__ __forceinline__ void tdm_load_2d(unsigned lds_off, const void* gptr,
                                                   unsigned tile_w, unsigned tile_h,
                                                   unsigned stride_elems)
{
    unsigned long long ga = (unsigned long long)(uintptr_t)gptr;
    u32x4 g0;
    g0[0] = 1u;                                   // count=1, user descriptor
    g0[1] = lds_off;                              // lds_addr (bytes)
    g0[2] = (unsigned)ga;                         // global_addr[31:0]
    g0[3] = (unsigned)(ga >> 32) | 0x80000000u;   // global_addr[56:32] | type=2<<30

    const unsigned dim0 = 0x7FFFFFFFu;            // effectively unbounded (no OOB clip)
    const unsigned dim1 = 0x7FFFFFFFu;
    i32x8 g1;
    g1[0] = (int)(1u << 16);                      // workgroup_mask=0, data_size=1 (2 bytes)
    g1[1] = (int)((dim0 & 0xFFFFu) << 16);        // tensor_dim0[15:0]
    g1[2] = (int)((dim0 >> 16) | ((dim1 & 0xFFFFu) << 16));  // dim0[31:16] | dim1[15:0]
    g1[3] = (int)((dim1 >> 16) | (tile_w << 16)); // dim1[31:16] | tile_dim0
    g1[4] = (int)tile_h;                          // tile_dim1 | tile_dim2=0
    g1[5] = (int)stride_elems;                    // tensor_dim0_stride[31:0]
    g1[6] = 0;                                    // stride0[47:32] | stride1[15:0]
    g1[7] = 0;                                    // stride1[47:16]

    i32x4 z4 = {0, 0, 0, 0};
#if TDM_6ARG
    i32x8 z8 = {0, 0, 0, 0, 0, 0, 0, 0};
    __builtin_amdgcn_tensor_load_to_lds(g0, g1, z4, z4, z8, 0);
#else
    __builtin_amdgcn_tensor_load_to_lds(g0, g1, z4, z4, 0);
#endif
}
#endif

// ---------------------------------------------------------------------------
// Generic bf16 GEMM:  C[M][N] = A[M][K](bf16,row-major) x Bt[K][N](bf16,row-major)
// 4 waves / block; each wave computes 32(M) x 64(N) so every B fragment feeds
// two WMMAs; block tile 128x64.  LDS double-buffered; tiles staged by the
// Tensor Data Mover (TENSORcnt), DMA of tile k+1 overlapped with WMMAs on k.
// ---------------------------------------------------------------------------
template<bool BIAS, bool RELU, bool RES, bool OUTF32, bool OUTB16, bool CAUSAL>
__global__ void __launch_bounds__(128)
gemm_bf16(const u16* __restrict__ A, int lda,
          const u16* __restrict__ Bt, int ldb,
          const float* __restrict__ bias,
          const float* __restrict__ res, int ldr,
          float* __restrict__ outf, int ldo,
          u16* __restrict__ outb, int ldob,
          int M, int N, int K)
{
    alignas(16) __shared__ u16 lA[2][128 * 32];  // [m][k] row-major, 8KB per buffer
    alignas(16) __shared__ u16 lB[2][32 * 64];   // [k][n] row-major, 4KB per buffer

    const int bm = blockIdx.y * 128;
    const int bn = blockIdx.x * 64;
    if (CAUSAL && bn >= bm + 128) return;         // fully above diagonal: skip block

    const int tid  = threadIdx.x;
    const int wave = tid >> 5;
    const int lane = tid & 31;
    const int lr   = lane & 15;
    const int hi   = lane >> 4;

    v8f z = {0,0,0,0,0,0,0,0};
    v8f acc00 = z, acc01 = z, acc02 = z, acc03 = z;   // M sub-tile 0
    v8f acc10 = z, acc11 = z, acc12 = z, acc13 = z;   // M sub-tile 1

#if HAVE_TDM
    auto stage = [&](int k0, int p) {
        if (tid < 32) {   // wave 0 issues the DMA; TDM ignores EXEC
            tdm_load_2d((unsigned)(uintptr_t)&lA[p][0], A + (size_t)bm * lda + k0,
                        32, 128, (unsigned)lda);
            tdm_load_2d((unsigned)(uintptr_t)&lB[p][0], Bt + (size_t)k0 * ldb + bn,
                        64, 32, (unsigned)ldb);
        }
    };
    auto commit = [&]() {
        if (tid < 32) __builtin_amdgcn_s_wait_tensorcnt(0);
        __syncthreads();
    };
#else
    auto stage = [&](int k0, int p) {
        {   // A tile: 128 rows x 32 elems; 1 thread/row, 64B each
            const int r = tid;
            const u16* g = A + (size_t)(bm + r) * lda + k0;
            *(v4u*)&lA[p][r * 32 +  0] = *(const v4u*)(g);
            *(v4u*)&lA[p][r * 32 +  8] = *(const v4u*)(g + 8);
            *(v4u*)&lA[p][r * 32 + 16] = *(const v4u*)(g + 16);
            *(v4u*)&lA[p][r * 32 + 24] = *(const v4u*)(g + 24);
        }
        {   // B tile: 32 rows x 64 elems; 4 threads/row, 16 elems each
            const int r = tid >> 2, seg = (tid & 3) * 16;
            const u16* g = Bt + (size_t)(k0 + r) * ldb + bn + seg;
            *(v4u*)&lB[p][r * 64 + seg]     = *(const v4u*)(g);
            *(v4u*)&lB[p][r * 64 + seg + 8] = *(const v4u*)(g + 8);
        }
    };
    auto commit = [&]() { __syncthreads(); };
#endif

    stage(0, 0);
    commit();

    int p = 0;
    for (int k0 = 0; k0 < K; k0 += 32) {
        if (k0 + 32 < K) stage(k0 + 32, p ^ 1);   // DMA next tile while computing this one

        // A frags: lane -> rows wave*32+lr (+16), K-offsets hi*8 + {0..7, 16..23}
        const u16* aL = &lA[p][(wave * 32 + lr) * 32 + hi * 8];
        Frag a0, a1;
        a0.u4[0] = *(const v4u*)(aL);
        a0.u4[1] = *(const v4u*)(aL + 16);
        a1.u4[0] = *(const v4u*)(aL + 16 * 32);
        a1.u4[1] = *(const v4u*)(aL + 16 * 32 + 16);
        // B frags: lane -> column N=lr (+16 per sub-tile), K-offsets hi*8 + {0..7, 16..23}
        const u16* bL = &lB[p][(hi * 8) * 64 + lr];
        Frag b0, b1, b2, b3;
        #pragma unroll
        for (int s = 0; s < 8; ++s) {
            ((u16*)&b0)[s] = bL[s * 64 +  0]; ((u16*)&b0)[8 + s] = bL[(16 + s) * 64 +  0];
            ((u16*)&b1)[s] = bL[s * 64 + 16]; ((u16*)&b1)[8 + s] = bL[(16 + s) * 64 + 16];
            ((u16*)&b2)[s] = bL[s * 64 + 32]; ((u16*)&b2)[8 + s] = bL[(16 + s) * 64 + 32];
            ((u16*)&b3)[s] = bL[s * 64 + 48]; ((u16*)&b3)[8 + s] = bL[(16 + s) * 64 + 48];
        }

        acc00 = __builtin_amdgcn_wmma_f32_16x16x32_bf16(false, a0.v, false, b0.v, (short)0, acc00, false, false);
        acc10 = __builtin_amdgcn_wmma_f32_16x16x32_bf16(false, a1.v, false, b0.v, (short)0, acc10, false, false);
        acc01 = __builtin_amdgcn_wmma_f32_16x16x32_bf16(false, a0.v, false, b1.v, (short)0, acc01, false, false);
        acc11 = __builtin_amdgcn_wmma_f32_16x16x32_bf16(false, a1.v, false, b1.v, (short)0, acc11, false, false);
        acc02 = __builtin_amdgcn_wmma_f32_16x16x32_bf16(false, a0.v, false, b2.v, (short)0, acc02, false, false);
        acc12 = __builtin_amdgcn_wmma_f32_16x16x32_bf16(false, a1.v, false, b2.v, (short)0, acc12, false, false);
        acc03 = __builtin_amdgcn_wmma_f32_16x16x32_bf16(false, a0.v, false, b3.v, (short)0, acc03, false, false);
        acc13 = __builtin_amdgcn_wmma_f32_16x16x32_bf16(false, a1.v, false, b3.v, (short)0, acc13, false, false);

        commit();
        p ^= 1;
    }

    // Epilogue.  C/D layout: element i of v8f on lane l -> M = m0 + i + 8*(l>=16), N = n0 + (l&15)
    const int n_lane = bn + lr;
    v8f accs[2][4] = {{acc00, acc01, acc02, acc03}, {acc10, acc11, acc12, acc13}};
    #pragma unroll
    for (int s = 0; s < 2; ++s) {
        const int m_base = bm + wave * 32 + s * 16 + hi * 8;
        #pragma unroll
        for (int t = 0; t < 4; ++t) {
            const int n = n_lane + t * 16;
            const float bv = BIAS ? bias[n] : 0.0f;
            #pragma unroll
            for (int i = 0; i < 8; ++i) {
                const int m = m_base + i;
                float v = accs[s][t][i] + bv;
                if (RES)  v += res[(size_t)m * ldr + n];
                if (RELU) v = v > 0.0f ? v : 0.0f;
                if (CAUSAL && n > m) v = -3.0e38f;
                if (OUTF32) outf[(size_t)m * ldo  + n] = v;
                if (OUTB16) outb[(size_t)m * ldob + n] = f2bf(v);
            }
        }
    }
}

// ---------------------------------------------------------------------------
// f32 [rows][cols] -> bf16 transposed [cols][rows]
// ---------------------------------------------------------------------------
__global__ void transpose_to_bf16(const float* __restrict__ src, u16* __restrict__ dst,
                                  int rows, int cols)
{
    int c = blockIdx.x * 16 + threadIdx.x;
    int r = blockIdx.y * 16 + threadIdx.y;
    if (r < rows && c < cols)
        dst[(size_t)c * rows + r] = f2bf(src[(size_t)r * cols + c]);
}

// ---------------------------------------------------------------------------
// RMSNorm per token: out(bf16) = x * rsqrt(mean(x^2)+eps) * w + b
// ---------------------------------------------------------------------------
__global__ void __launch_bounds__(256)
rmsnorm_kernel(const float* __restrict__ x, const float* __restrict__ w,
               const float* __restrict__ b, u16* __restrict__ out)
{
    const int s = blockIdx.x;
    __shared__ float red[256];
    const float* xr = x + (size_t)s * D_MODEL;
    float ss = 0.0f;
    for (int i = threadIdx.x; i < D_MODEL; i += 256) { float v = xr[i]; ss += v * v; }
    red[threadIdx.x] = ss; __syncthreads();
    for (int o = 128; o > 0; o >>= 1) {
        if (threadIdx.x < o) red[threadIdx.x] += red[threadIdx.x + o];
        __syncthreads();
    }
    const float inv = rsqrtf(red[0] * (1.0f / D_MODEL) + 1e-5f);
    for (int i = threadIdx.x; i < D_MODEL; i += 256)
        out[(size_t)s * D_MODEL + i] = f2bf(xr[i] * inv * w[i] + b[i]);
}

// ---------------------------------------------------------------------------
// RoPE (rotate-half / NeoX).  q -> qb (bf16, scaled by 1/sqrt(HD));
// k -> kbt (bf16, layout [h][d][s] so score-GEMM B operand is contiguous).
// ---------------------------------------------------------------------------
__global__ void __launch_bounds__(256)
rope_kernel(const float* __restrict__ qf, const float* __restrict__ kf,
            u16* __restrict__ qb, u16* __restrict__ kbt)
{
    const int s = blockIdx.x;
    const int idx = blockIdx.y * 256 + threadIdx.x;    // 0..1023
    const int h = idx >> 6;
    const int d = idx & 63;
    const int j = d & 31;
    const float ang = (float)s * expf(-(float)j * (9.210340371976184f / 32.0f)); // theta^(-2j/64)
    const float c = cosf(ang), si = sinf(ang);
    const int partner = h * HD + ((d < 32) ? d + 32 : d - 32);
    {
        float x = qf[(size_t)s * D_MODEL + idx];
        float o = qf[(size_t)s * D_MODEL + partner];
        float rot = (d < 32) ? -o : o;
        qb[(size_t)s * D_MODEL + idx] = f2bf((x * c + rot * si) * 0.125f); // fold 1/sqrt(64)
    }
    {
        float x = kf[(size_t)s * D_MODEL + idx];
        float o = kf[(size_t)s * D_MODEL + partner];
        float rot = (d < 32) ? -o : o;
        kbt[((size_t)h * HD + d) * SEQ + s] = f2bf(x * c + rot * si);
    }
}

// ---------------------------------------------------------------------------
// Causal softmax over scores row s (cols 0..s valid); writes bf16 probs,
// zeros above the diagonal (upper score blocks were never written).
// ---------------------------------------------------------------------------
__global__ void __launch_bounds__(256)
softmax_kernel(const float* __restrict__ scores, u16* __restrict__ probs)
{
    const int s = blockIdx.x;
    const int n = s + 1;
    __shared__ float red[256];
    const float* row = scores + (size_t)s * SEQ;

    float mx = -3.4e38f;
    for (int j = threadIdx.x; j < n; j += 256) mx = fmaxf(mx, row[j]);
    red[threadIdx.x] = mx; __syncthreads();
    for (int o = 128; o > 0; o >>= 1) {
        if (threadIdx.x < o) red[threadIdx.x] = fmaxf(red[threadIdx.x], red[threadIdx.x + o]);
        __syncthreads();
    }
    const float m = red[0]; __syncthreads();

    float sum = 0.0f;
    for (int j = threadIdx.x; j < n; j += 256) sum += __expf(row[j] - m);
    red[threadIdx.x] = sum; __syncthreads();
    for (int o = 128; o > 0; o >>= 1) {
        if (threadIdx.x < o) red[threadIdx.x] += red[threadIdx.x + o];
        __syncthreads();
    }
    const float invs = 1.0f / red[0];

    for (int j = threadIdx.x; j < SEQ; j += 256)
        probs[(size_t)s * SEQ + j] = (j < n) ? f2bf(__expf(row[j] - m) * invs) : (u16)0;
}

// ---------------------------------------------------------------------------
// Convs (tiny channel counts, scalar f32)
// ---------------------------------------------------------------------------
__global__ void __launch_bounds__(256)
conv_in_kernel(const float* __restrict__ x, const float* __restrict__ w,
               const float* __restrict__ b, float* __restrict__ h)
{
    const int t = blockIdx.x * 256 + threadIdx.x;  // 0..2047
    const int c = blockIdx.y;                      // 0..1023
    float acc = b[c];
    #pragma unroll
    for (int ci = 0; ci < 4; ++ci)
        #pragma unroll
        for (int k = 0; k < KCONV; ++k) {
            const int tt = t + k - (KCONV - 1);
            if (tt >= 0) acc += w[(c * 4 + ci) * KCONV + k] * x[ci * SEQ + tt];
        }
    h[(size_t)t * D_MODEL + c] = acc > 0.0f ? acc : 0.0f;   // relu, transposed -> [L][D]
}

__global__ void __launch_bounds__(256)
conv_out_kernel(const float* __restrict__ h, const float* __restrict__ w,
                const float* __restrict__ b, float* __restrict__ out)
{
    const int t = blockIdx.x * 256 + threadIdx.x;  // 0..2047
    const int co = blockIdx.y;                     // 0..3
    float acc = b[co];
    for (int ci = 0; ci < D_MODEL; ++ci) {
        const float* wr = w + (size_t)(co * D_MODEL + ci) * KCONV;
        #pragma unroll
        for (int k = 0; k < KCONV; ++k) {
            const int tt = t + k - (KCONV - 1);
            if (tt >= 0) acc += wr[k] * h[(size_t)tt * D_MODEL + ci];
        }
    }
    out[(size_t)co * SEQ + t] = acc;
}

// ---------------------------------------------------------------------------
// Host launcher
// ---------------------------------------------------------------------------
extern "C" void kernel_launch(void* const* d_in, const int* in_sizes, int n_in,
                              void* d_out, int out_size, void* d_ws, size_t ws_size,
                              hipStream_t stream)
{
    const float* x_in      = (const float*)d_in[0];
    const float* conv_in_w = (const float*)d_in[1];
    const float* conv_in_b = (const float*)d_in[2];
    const float* ln1_w     = (const float*)d_in[3];
    const float* ln1_b     = (const float*)d_in[4];
    const float* wq        = (const float*)d_in[5];
    const float* wk        = (const float*)d_in[6];
    const float* wv        = (const float*)d_in[7];
    const float* wo        = (const float*)d_in[8];
    const float* ln2_w     = (const float*)d_in[9];
    const float* ln2_b     = (const float*)d_in[10];
    const float* w1        = (const float*)d_in[11];
    const float* b1        = (const float*)d_in[12];
    const float* w2        = (const float*)d_in[13];
    const float* b2        = (const float*)d_in[14];
    const float* w3        = (const float*)d_in[15];
    const float* b3        = (const float*)d_in[16];
    const float* conv_out_w= (const float*)d_in[17];
    const float* conv_out_b= (const float*)d_in[18];

    char* wp = (char*)d_ws;
    auto alloc = [&](size_t bytes) -> void* {
        void* r = (void*)wp; wp += (bytes + 255) & ~(size_t)255; return r;
    };
    float* h      = (float*)alloc((size_t)SEQ * D_MODEL * 4);
    u16*   hn     = (u16*)  alloc((size_t)SEQ * D_MODEL * 2);
    float* qf     = (float*)alloc((size_t)SEQ * D_MODEL * 4);
    float* kf     = (float*)alloc((size_t)SEQ * D_MODEL * 4);
    u16*   qb     = (u16*)  alloc((size_t)SEQ * D_MODEL * 2);
    u16*   kbt    = (u16*)  alloc((size_t)SEQ * D_MODEL * 2);   // [h][d][s]
    u16*   vb     = (u16*)  alloc((size_t)SEQ * D_MODEL * 2);
    u16*   ob     = (u16*)  alloc((size_t)SEQ * D_MODEL * 2);
    float* scores = (float*)alloc((size_t)SEQ * SEQ * 4);
    u16*   probs  = (u16*)  alloc((size_t)SEQ * SEQ * 2);
    u16*   h1     = (u16*)  alloc((size_t)SEQ * D_FF * 2);
    u16*   h2     = (u16*)  alloc((size_t)SEQ * D_FF * 2);
    u16*   wqt    = (u16*)  alloc((size_t)D_MODEL * D_MODEL * 2);
    u16*   wkt    = (u16*)  alloc((size_t)D_MODEL * D_MODEL * 2);
    u16*   wvt    = (u16*)  alloc((size_t)D_MODEL * D_MODEL * 2);
    u16*   wot    = (u16*)  alloc((size_t)D_MODEL * D_MODEL * 2);
    u16*   w1t    = (u16*)  alloc((size_t)D_MODEL * D_FF * 2);
    u16*   w2t    = (u16*)  alloc((size_t)D_FF * D_FF * 2);
    u16*   w3t    = (u16*)  alloc((size_t)D_FF * D_MODEL * 2);

    const dim3 blk128(128), blk256(256), blkT(16, 16);

    conv_in_kernel<<<dim3(SEQ / 256, D_MODEL), blk256, 0, stream>>>(x_in, conv_in_w, conv_in_b, h);

    for (int l = 0; l < NUM_LAYERS; ++l) {
        // per-layer weight conversion (f32 -> bf16, transposed to [K][N])
        transpose_to_bf16<<<dim3(D_MODEL/16, D_MODEL/16), blkT, 0, stream>>>(wq + (size_t)l*D_MODEL*D_MODEL, wqt, D_MODEL, D_MODEL);
        transpose_to_bf16<<<dim3(D_MODEL/16, D_MODEL/16), blkT, 0, stream>>>(wk + (size_t)l*D_MODEL*D_MODEL, wkt, D_MODEL, D_MODEL);
        transpose_to_bf16<<<dim3(D_MODEL/16, D_MODEL/16), blkT, 0, stream>>>(wv + (size_t)l*D_MODEL*D_MODEL, wvt, D_MODEL, D_MODEL);
        transpose_to_bf16<<<dim3(D_MODEL/16, D_MODEL/16), blkT, 0, stream>>>(wo + (size_t)l*D_MODEL*D_MODEL, wot, D_MODEL, D_MODEL);
        transpose_to_bf16<<<dim3(D_MODEL/16, D_FF/16),    blkT, 0, stream>>>(w1 + (size_t)l*D_FF*D_MODEL,   w1t, D_FF,   D_MODEL);
        transpose_to_bf16<<<dim3(D_FF/16,    D_FF/16),    blkT, 0, stream>>>(w2 + (size_t)l*D_FF*D_FF,      w2t, D_FF,   D_FF);
        transpose_to_bf16<<<dim3(D_FF/16,    D_MODEL/16), blkT, 0, stream>>>(w3 + (size_t)l*D_MODEL*D_FF,   w3t, D_MODEL, D_FF);

        // ---- attention ----
        rmsnorm_kernel<<<SEQ, blk256, 0, stream>>>(h, ln1_w + l*D_MODEL, ln1_b + l*D_MODEL, hn);

        gemm_bf16<false,false,false,true,false,false><<<dim3(D_MODEL/64, SEQ/128), blk128, 0, stream>>>(
            hn, D_MODEL, wqt, D_MODEL, nullptr, nullptr, 0, qf, D_MODEL, nullptr, 0, SEQ, D_MODEL, D_MODEL);
        gemm_bf16<false,false,false,true,false,false><<<dim3(D_MODEL/64, SEQ/128), blk128, 0, stream>>>(
            hn, D_MODEL, wkt, D_MODEL, nullptr, nullptr, 0, kf, D_MODEL, nullptr, 0, SEQ, D_MODEL, D_MODEL);
        gemm_bf16<false,false,false,false,true,false><<<dim3(D_MODEL/64, SEQ/128), blk128, 0, stream>>>(
            hn, D_MODEL, wvt, D_MODEL, nullptr, nullptr, 0, nullptr, 0, vb, D_MODEL, SEQ, D_MODEL, D_MODEL);

        rope_kernel<<<dim3(SEQ, D_MODEL/256), blk256, 0, stream>>>(qf, kf, qb, kbt);

        for (int hh = 0; hh < N_HEADS; ++hh) {
            // scores = (q/sqrt(hd)) @ k^T, causal
            gemm_bf16<false,false,false,true,false,true><<<dim3(SEQ/64, SEQ/128), blk128, 0, stream>>>(
                qb + hh*HD, D_MODEL, kbt + (size_t)hh*HD*SEQ, SEQ,
                nullptr, nullptr, 0, scores, SEQ, nullptr, 0, SEQ, SEQ, HD);
            softmax_kernel<<<SEQ, blk256, 0, stream>>>(scores, probs);
            // o = probs @ v
            gemm_bf16<false,false,false,false,true,false><<<dim3(HD/64, SEQ/128), blk128, 0, stream>>>(
                probs, SEQ, vb + hh*HD, D_MODEL,
                nullptr, nullptr, 0, nullptr, 0, ob + hh*HD, D_MODEL, SEQ, HD, SEQ);
        }

        // x = x + o @ wo^T   (residual fused, in-place on h)
        gemm_bf16<false,false,true,true,false,false><<<dim3(D_MODEL/64, SEQ/128), blk128, 0, stream>>>(
            ob, D_MODEL, wot, D_MODEL, nullptr, h, D_MODEL, h, D_MODEL, nullptr, 0, SEQ, D_MODEL, D_MODEL);

        // ---- MLP ----
        rmsnorm_kernel<<<SEQ, blk256, 0, stream>>>(h, ln2_w + l*D_MODEL, ln2_b + l*D_MODEL, hn);

        gemm_bf16<true,true,false,false,true,false><<<dim3(D_FF/64, SEQ/128), blk128, 0, stream>>>(
            hn, D_MODEL, w1t, D_FF, b1 + l*D_FF, nullptr, 0, nullptr, 0, h1, D_FF, SEQ, D_FF, D_MODEL);
        gemm_bf16<true,true,false,false,true,false><<<dim3(D_FF/64, SEQ/128), blk128, 0, stream>>>(
            h1, D_FF, w2t, D_FF, b2 + l*D_FF, nullptr, 0, nullptr, 0, h2, D_FF, SEQ, D_FF, D_FF);
        gemm_bf16<true,false,true,true,false,false><<<dim3(D_MODEL/64, SEQ/128), blk128, 0, stream>>>(
            h2, D_FF, w3t, D_MODEL, b3 + l*D_MODEL, h, D_MODEL, h, D_MODEL, nullptr, 0, SEQ, D_MODEL, D_FF);
    }

    conv_out_kernel<<<dim3(SEQ / 256, 4), blk256, 0, stream>>>(h, conv_out_w, conv_out_b, (float*)d_out);
}
